// BiMultiHeadAttention_79783312491083
// MI455X (gfx1250) — compile-verified, matrix operand order checked by reading, not run on hardware
//
#include <hip/hip_runtime.h>

// ---------------------------------------------------------------------------
// BiMultiHeadAttention for MI455X (gfx1250, wave32, WMMA).
// All GEMMs run through v_wmma_f32_16x16x32_bf16 (bf16 in, fp32 accumulate),
// with double-buffered LDS staging and global_prefetch_b8 look-ahead.
// ---------------------------------------------------------------------------

#define B_   4
#define NV_  4096
#define NL_  256
#define VD_  1024
#define LD_  768
#define E_   1024
#define H_   16
#define D_   64          // head dim
#define SCALE_ 0.25f     // HEADS ** -0.5

typedef __bf16 bf16;
typedef __attribute__((ext_vector_type(16))) __bf16 v16bf;
typedef __attribute__((ext_vector_type(8)))  __bf16 v8bf;
typedef __attribute__((ext_vector_type(4)))  __bf16 v4bf;
typedef __attribute__((ext_vector_type(8)))  float  v8f;

// ---------------------------------------------------------------- convert ---
__global__ __launch_bounds__(256)
void k_f32_to_bf16(const float* __restrict__ x, bf16* __restrict__ y, long n) {
  long i = ((long)blockIdx.x * 256 + threadIdx.x) * 4;
  if (i < n) {
    float4 f = *(const float4*)(x + i);
    v4bf o;
    o[0] = (bf16)f.x; o[1] = (bf16)f.y; o[2] = (bf16)f.z; o[3] = (bf16)f.w;
    *(v4bf*)(y + i) = o;
  }
}

// --------------------------------------------- weight transpose f32->bf16 ---
// W [K][N] (row-major, [in,out]) -> Wt [N][K] bf16 so GEMM B-operand rows are
// contiguous along K (matches WMMA per-lane K-striped fragment layout).
__global__ __launch_bounds__(256)
void k_wT(const float* __restrict__ W, bf16* __restrict__ Wt, int K, int N) {
  __shared__ float tile[32][33];
  int tx = threadIdx.x, ty = threadIdx.y;
  int n0 = blockIdx.x * 32, k0 = blockIdx.y * 32;
  #pragma unroll
  for (int j = 0; j < 32; j += 8)
    tile[ty + j][tx] = W[(long)(k0 + ty + j) * N + n0 + tx];
  __syncthreads();
  #pragma unroll
  for (int j = 0; j < 32; j += 8)
    Wt[(long)(n0 + ty + j) * K + k0 + tx] = (bf16)tile[tx][ty + j];
}

// ------------------------------------------------ per-head V transpose ------
// X [B, N, H*D] bf16  ->  XT [B, H, D, N] bf16   (contiguous K for PV GEMMs)
__global__ __launch_bounds__(256)
void k_headT(const bf16* __restrict__ X, bf16* __restrict__ XT, int N) {
  __shared__ bf16 tile[32][33];
  int tx = threadIdx.x, ty = threadIdx.y;
  int b = blockIdx.z / H_, h = blockIdx.z % H_;
  int n0 = blockIdx.x * 32, d0 = blockIdx.y * 32;
  #pragma unroll
  for (int j = 0; j < 32; j += 8)
    tile[ty + j][tx] = X[((long)(b * N + n0 + ty + j)) * E_ + h * D_ + d0 + tx];
  __syncthreads();
  #pragma unroll
  for (int j = 0; j < 32; j += 8)
    XT[((long)((b * H_ + h) * D_ + d0 + ty + j)) * N + n0 + tx] = tile[tx][ty + j];
}

// ---------------------------------------------------------------------------
// Batched WMMA GEMM: C = scale * (A @ B^T-layout) [+ bias]
//   A  : [M,K] bf16, row stride lda, contiguous K
//   Bm : [N,K] bf16, row stride ldb, contiguous K
//   C  : [M,N] (float or bf16), row stride ldc
// grid.z = batch; z -> (zb = z/Hn, zh = z%Hn); per-axis strides sXb/sXh.
// Block 256 threads = 8 waves; 128x128 tile, K-step 32 (== WMMA K).
// Each wave: 32x64 sub-tile = 2x4 accumulators of v8f.
// FULL: M%128==0 && N%128==0 at launch -> all bounds checks compiled out.
// Double-buffered LDS: next tile's global loads overlap current WMMA work.
// ---------------------------------------------------------------------------
#define BM 128
#define BN 128
#define BK 32
#define LSTR 40   // LDS row stride in bf16 (80B: 16B aligned, conflict-free)

template <typename CT, bool BIAS, bool FULL>
__global__ __launch_bounds__(256)
void k_gemm(const bf16* __restrict__ A, const bf16* __restrict__ Bm,
            const float* __restrict__ bias, CT* __restrict__ C,
            int M, int N, int K, int lda, int ldb, int ldc,
            long long sAb, long long sAh, long long sBb, long long sBh,
            long long sCb, long long sCh, int Hn, float scale) {
  __shared__ __align__(16) bf16 As[2 * BM * LSTR];
  __shared__ __align__(16) bf16 Bs[2 * BN * LSTR];

  int zb = blockIdx.z / Hn, zh = blockIdx.z % Hn;
  A  += (long long)zb * sAb + (long long)zh * sAh;
  Bm += (long long)zb * sBb + (long long)zh * sBh;
  C  += (long long)zb * sCb + (long long)zh * sCh;

  int t = threadIdx.x, lane = t & 31, wid = t >> 5;
  int bm = blockIdx.y * BM, bn = blockIdx.x * BN;
  int wm = (wid & 3) * 32, wn = (wid >> 2) * 64;

  // Per-thread staging slots: rows r0 and r0+64, 8 contiguous K at column c0.
  int r0 = t >> 2, c0 = (t & 3) * 8;
  const bf16* Apt0 = A + (long long)(bm + r0) * lda + c0;
  const bf16* Apt1 = A + (long long)(bm + r0 + 64) * lda + c0;
  const bf16* Bpt0 = Bm + (long long)(bn + r0) * ldb + c0;
  const bf16* Bpt1 = Bm + (long long)(bn + r0 + 64) * ldb + c0;
  bool gA0 = FULL || (bm + r0 < M),      gA1 = FULL || (bm + r0 + 64 < M);
  bool gB0 = FULL || (bn + r0 < N),      gB1 = FULL || (bn + r0 + 64 < N);

  v8f zf = {};
  v8f acc[2][4];
  #pragma unroll
  for (int mt = 0; mt < 2; ++mt)
    #pragma unroll
    for (int nt = 0; nt < 4; ++nt) acc[mt][nt] = zf;

  v8bf zb8 = {};
  int rlane = lane & 15;
  int ksel  = (lane >> 4) * 8;  // lanes 0-15: K {0..7,16..23}; 16-31: {8..15,24..31}

  // ---- prologue: stage tile k0=0 into buffer 0 ----
  {
    v8bf a0 = gA0 ? *(const v8bf*)Apt0 : zb8;
    v8bf a1 = gA1 ? *(const v8bf*)Apt1 : zb8;
    v8bf b0 = gB0 ? *(const v8bf*)Bpt0 : zb8;
    v8bf b1 = gB1 ? *(const v8bf*)Bpt1 : zb8;
    *(v8bf*)(As + r0 * LSTR + c0)        = a0;
    *(v8bf*)(As + (r0 + 64) * LSTR + c0) = a1;
    *(v8bf*)(Bs + r0 * LSTR + c0)        = b0;
    *(v8bf*)(Bs + (r0 + 64) * LSTR + c0) = b1;
  }
  __syncthreads();

  int buf = 0;
  for (int k0 = 0; k0 < K; k0 += BK) {
    bool notlast = (k0 + BK < K);

    // ---- issue next tile's global loads (latency hidden under WMMA) ----
    v8bf na0 = zb8, na1 = zb8, nb0 = zb8, nb1 = zb8;
    if (notlast) {
      int kn = k0 + BK;
      if (gA0) na0 = *(const v8bf*)(Apt0 + kn);
      if (gA1) na1 = *(const v8bf*)(Apt1 + kn);
      if (gB0) nb0 = *(const v8bf*)(Bpt0 + kn);
      if (gB1) nb1 = *(const v8bf*)(Bpt1 + kn);
      if (k0 + 2 * BK < K) {   // pre-warm L2/WGP$ two steps ahead
        __builtin_prefetch(Apt0 + k0 + 2 * BK, 0, 1);
        __builtin_prefetch(Bpt0 + k0 + 2 * BK, 0, 1);
      }
    }

    // ---- compute on current buffer ----
    const bf16* Ab = As + buf * (BM * LSTR);
    const bf16* Bb = Bs + buf * (BN * LSTR);
    v16bf af[2], bfx[4];
    #pragma unroll
    for (int mt = 0; mt < 2; ++mt) {
      int row = wm + mt * 16 + rlane;
      v8bf lo = *(const v8bf*)(Ab + row * LSTR + ksel);
      v8bf hi = *(const v8bf*)(Ab + row * LSTR + 16 + ksel);
      af[mt] = __builtin_shufflevector(lo, hi, 0,1,2,3,4,5,6,7,8,9,10,11,12,13,14,15);
    }
    #pragma unroll
    for (int nt = 0; nt < 4; ++nt) {
      int row = wn + nt * 16 + rlane;
      v8bf lo = *(const v8bf*)(Bb + row * LSTR + ksel);
      v8bf hi = *(const v8bf*)(Bb + row * LSTR + 16 + ksel);
      bfx[nt] = __builtin_shufflevector(lo, hi, 0,1,2,3,4,5,6,7,8,9,10,11,12,13,14,15);
    }
    #pragma unroll
    for (int mt = 0; mt < 2; ++mt)
      #pragma unroll
      for (int nt = 0; nt < 4; ++nt)
        acc[mt][nt] = __builtin_amdgcn_wmma_f32_16x16x32_bf16(
            false, af[mt], false, bfx[nt], (short)0, acc[mt][nt], false, false);

    // ---- stage next tile into alternate buffer ----
    if (notlast) {
      bf16* Ad = As + (buf ^ 1) * (BM * LSTR);
      bf16* Bd = Bs + (buf ^ 1) * (BN * LSTR);
      *(v8bf*)(Ad + r0 * LSTR + c0)        = na0;
      *(v8bf*)(Ad + (r0 + 64) * LSTR + c0) = na1;
      *(v8bf*)(Bd + r0 * LSTR + c0)        = nb0;
      *(v8bf*)(Bd + (r0 + 64) * LSTR + c0) = nb1;
    }
    __syncthreads();
    buf ^= 1;
  }

  // Store per ISA C/D layout: VGPR g, lane L -> m = g + (L>=16 ? 8 : 0), n = L&15.
  #pragma unroll
  for (int mt = 0; mt < 2; ++mt)
    #pragma unroll
    for (int nt = 0; nt < 4; ++nt)
      #pragma unroll
      for (int g = 0; g < 8; ++g) {
        int m = bm + wm + mt * 16 + g + ((lane >> 4) * 8);
        int n = bn + wn + nt * 16 + (lane & 15);
        if (FULL || (m < M && n < N)) {
          float cv = acc[mt][nt][g] * scale;
          if (BIAS) cv += bias[n];
          C[(long long)m * ldc + n] = (CT)cv;
        }
      }
}

// ----------------------------------------------- row softmax (over Nl) ------
// scores [B,H,Nv,Nl] f32 (+mask bias) -> attn_v bf16. One wave per row.
__global__ __launch_bounds__(256)
void k_softmax_rows(const float* __restrict__ S, const int* __restrict__ maskL,
                    bf16* __restrict__ out) {
  int wid = threadIdx.x >> 5, lane = threadIdx.x & 31;
  long row = (long)blockIdx.x * 8 + wid;
  int b = (int)(row / ((long)H_ * NV_));
  const float* p = S + row * NL_;
  float v[8];
  float4 x0 = *(const float4*)(p + lane * 8);
  float4 x1 = *(const float4*)(p + lane * 8 + 4);
  v[0]=x0.x; v[1]=x0.y; v[2]=x0.z; v[3]=x0.w;
  v[4]=x1.x; v[5]=x1.y; v[6]=x1.z; v[7]=x1.w;
  #pragma unroll
  for (int i = 0; i < 8; ++i) {
    int col = lane * 8 + i;
    if (maskL[b * NL_ + col] == 0) v[i] += -1e9f;
  }
  float m = v[0];
  #pragma unroll
  for (int i = 1; i < 8; ++i) m = fmaxf(m, v[i]);
  for (int o = 16; o; o >>= 1) m = fmaxf(m, __shfl_xor(m, o));
  float s = 0.f;
  #pragma unroll
  for (int i = 0; i < 8; ++i) { v[i] = __expf(v[i] - m); s += v[i]; }
  for (int o = 16; o; o >>= 1) s += __shfl_xor(s, o);
  float inv = 1.f / s;
  v8bf o8;
  #pragma unroll
  for (int i = 0; i < 8; ++i) o8[i] = (bf16)(v[i] * inv);
  *(v8bf*)(out + row * NL_ + lane * 8) = o8;
}

// --------------------------------------- column softmax (over Nv), transp ---
// scores [B,H,Nv,Nl] f32 -> attn_l^T [B,H,Nl,Nv] bf16. One block per column.
// Strided score reads stay L2-resident (4MB per (b,h) slice; scores 256MB in total).
__global__ __launch_bounds__(256)
void k_softmax_cols(const float* __restrict__ S, bf16* __restrict__ outT) {
  __shared__ float red[256];
  int t = threadIdx.x;
  int nl = blockIdx.x, h = blockIdx.y, b = blockIdx.z;
  long base = ((long)(b * H_ + h) * NV_) * NL_ + nl;
  float v[16];
  float m = -3.4e38f;
  #pragma unroll
  for (int i = 0; i < 16; ++i) {
    v[i] = S[base + (long)(t + i * 256) * NL_];
    m = fmaxf(m, v[i]);
  }
  red[t] = m; __syncthreads();
  for (int s = 128; s > 0; s >>= 1) {
    if (t < s) red[t] = fmaxf(red[t], red[t + s]);
    __syncthreads();
  }
  m = red[0]; __syncthreads();
  float sum = 0.f;
  #pragma unroll
  for (int i = 0; i < 16; ++i) { v[i] = __expf(v[i] - m); sum += v[i]; }
  red[t] = sum; __syncthreads();
  for (int s = 128; s > 0; s >>= 1) {
    if (t < s) red[t] += red[t + s];
    __syncthreads();
  }
  float inv = 1.f / red[0];
  long obase = ((long)(b * H_ + h) * NL_ + nl) * NV_;
  #pragma unroll
  for (int i = 0; i < 16; ++i)
    outT[obase + t + i * 256] = (bf16)(v[i] * inv);
}

// ---------------------------------------------------------------------------
extern "C" void kernel_launch(void* const* d_in, const int* in_sizes, int n_in,
                              void* d_out, int out_size, void* d_ws, size_t ws_size,
                              hipStream_t stream) {
  (void)in_sizes; (void)n_in; (void)out_size; (void)ws_size;

  const float* v       = (const float*)d_in[0];
  const float* l       = (const float*)d_in[1];
  const int*   maskL   = (const int*)d_in[2];
  const float* W_v2q   = (const float*)d_in[3];  const float* b_v2q  = (const float*)d_in[4];
  const float* W_l2k   = (const float*)d_in[5];  const float* b_l2k  = (const float*)d_in[6];
  const float* W_v2v   = (const float*)d_in[7];  const float* b_v2v  = (const float*)d_in[8];
  const float* W_l2v   = (const float*)d_in[9];  const float* b_l2v  = (const float*)d_in[10];
  const float* W_v2out = (const float*)d_in[11]; const float* b_v2o  = (const float*)d_in[12];
  const float* W_l2out = (const float*)d_in[13]; const float* b_l2o  = (const float*)d_in[14];

  float* outV = (float*)d_out;
  float* outL = outV + (long long)B_ * NV_ * VD_;

  // --- bump allocator over workspace (256B aligned) ---
  char* ws = (char*)d_ws; size_t off = 0;
  auto alloc = [&](size_t bytes) -> void* {
    void* p = ws + off; off = (off + bytes + 255) & ~(size_t)255; return p;
  };
  bf16*  vB      = (bf16*)alloc((size_t)B_ * NV_ * VD_ * 2);
  bf16*  lB      = (bf16*)alloc((size_t)B_ * NL_ * LD_ * 2);
  bf16*  Wt_v2q  = (bf16*)alloc((size_t)E_ * VD_ * 2);
  bf16*  Wt_l2k  = (bf16*)alloc((size_t)E_ * LD_ * 2);
  bf16*  Wt_v2v  = (bf16*)alloc((size_t)E_ * VD_ * 2);
  bf16*  Wt_l2v  = (bf16*)alloc((size_t)E_ * LD_ * 2);
  bf16*  Wt_v2o  = (bf16*)alloc((size_t)VD_ * E_ * 2);
  bf16*  Wt_l2o  = (bf16*)alloc((size_t)LD_ * E_ * 2);
  bf16*  qB      = (bf16*)alloc((size_t)B_ * NV_ * E_ * 2);
  bf16*  kB      = (bf16*)alloc((size_t)B_ * NL_ * E_ * 2);
  bf16*  valvB   = (bf16*)alloc((size_t)B_ * NV_ * E_ * 2);
  bf16*  vallB   = (bf16*)alloc((size_t)B_ * NL_ * E_ * 2);
  bf16*  vallT   = (bf16*)alloc((size_t)B_ * H_ * D_ * NL_ * 2);
  float* scoresF = (float*)alloc((size_t)B_ * H_ * NV_ * NL_ * 4);
  bf16*  attnV   = (bf16*)alloc((size_t)B_ * H_ * NV_ * NL_ * 2);
  bf16*  attnLT  = (bf16*)alloc((size_t)B_ * H_ * NL_ * NV_ * 2);
  bf16*  olB     = (bf16*)alloc((size_t)B_ * NL_ * E_ * 2);
  bf16*  valvT   = qB;   // q is dead after the score GEMM (same size)
  bf16*  ovB     = vB;   // v_bf16 is dead after the q/valv projections

  dim3 b256(256), bT(32, 8);

  // 1) fp32 -> bf16 activations
  k_f32_to_bf16<<<(B_*NV_*VD_/4 + 255)/256, b256, 0, stream>>>(v, vB, (long)B_*NV_*VD_);
  k_f32_to_bf16<<<(B_*NL_*LD_/4 + 255)/256, b256, 0, stream>>>(l, lB, (long)B_*NL_*LD_);

  // 2) transpose weights to [out][in] bf16
  k_wT<<<dim3(E_/32,  VD_/32), bT, 0, stream>>>(W_v2q,   Wt_v2q, VD_, E_);
  k_wT<<<dim3(E_/32,  LD_/32), bT, 0, stream>>>(W_l2k,   Wt_l2k, LD_, E_);
  k_wT<<<dim3(E_/32,  VD_/32), bT, 0, stream>>>(W_v2v,   Wt_v2v, VD_, E_);
  k_wT<<<dim3(E_/32,  LD_/32), bT, 0, stream>>>(W_l2v,   Wt_l2v, LD_, E_);
  k_wT<<<dim3(VD_/32, E_/32),  bT, 0, stream>>>(W_v2out, Wt_v2o, E_, VD_);
  k_wT<<<dim3(LD_/32, E_/32),  bT, 0, stream>>>(W_l2out, Wt_l2o, E_, LD_);

  // 3) input projections (bf16 out, +bias) — full tiles, no guards
  k_gemm<bf16, true, true><<<dim3(E_/128, (B_*NV_)/128, 1), b256, 0, stream>>>(
      vB, Wt_v2q, b_v2q, qB, B_*NV_, E_, VD_, VD_, VD_, E_, 0,0,0,0,0,0, 1, 1.0f);
  k_gemm<bf16, true, true><<<dim3(E_/128, (B_*NL_)/128, 1), b256, 0, stream>>>(
      lB, Wt_l2k, b_l2k, kB, B_*NL_, E_, LD_, LD_, LD_, E_, 0,0,0,0,0,0, 1, 1.0f);
  k_gemm<bf16, true, true><<<dim3(E_/128, (B_*NV_)/128, 1), b256, 0, stream>>>(
      vB, Wt_v2v, b_v2v, valvB, B_*NV_, E_, VD_, VD_, VD_, E_, 0,0,0,0,0,0, 1, 1.0f);
  k_gemm<bf16, true, true><<<dim3(E_/128, (B_*NL_)/128, 1), b256, 0, stream>>>(
      lB, Wt_l2v, b_l2v, vallB, B_*NL_, E_, LD_, LD_, LD_, E_, 0,0,0,0,0,0, 1, 1.0f);

  // 4) scores = scale * Q·K^T per (b,h): fp32 out — full tiles
  k_gemm<float, false, true><<<dim3(NL_/128, NV_/128, B_*H_), b256, 0, stream>>>(
      qB, kB, nullptr, scoresF, NV_, NL_, D_, E_, E_, NL_,
      (long long)NV_*E_, D_, (long long)NL_*E_, D_,
      (long long)H_*NV_*NL_, (long long)NV_*NL_, H_, SCALE_);

  // 5) softmaxes (row: +mask; col: writes attn_l transposed)
  k_softmax_rows<<<B_*H_*NV_/8, b256, 0, stream>>>(scoresF, maskL, attnV);
  k_softmax_cols<<<dim3(NL_, H_, B_), b256, 0, stream>>>(scoresF, attnLT);

  // 6) per-head transposes of V matrices
  k_headT<<<dim3(NL_/32, D_/32, B_*H_), bT, 0, stream>>>(vallB, vallT, NL_);
  k_headT<<<dim3(NV_/32, D_/32, B_*H_), bT, 0, stream>>>(valvB, valvT, NV_);

  // 7) out_v = attn_v @ val_l   (per-head, N=64 -> guarded tiles)
  k_gemm<bf16, false, false><<<dim3(1, NV_/128, B_*H_), b256, 0, stream>>>(
      attnV, vallT, nullptr, ovB, NV_, D_, NL_, NL_, NL_, E_,
      (long long)H_*NV_*NL_, (long long)NV_*NL_,
      (long long)H_*D_*NL_, (long long)D_*NL_,
      (long long)NV_*E_, D_, H_, 1.0f);

  // 8) out_l = attn_l^T @ val_v (per-head, K=Nv, N=64 -> guarded tiles)
  k_gemm<bf16, false, false><<<dim3(1, NL_/128, B_*H_), b256, 0, stream>>>(
      attnLT, valvT, nullptr, olB, NL_, D_, NV_, NV_, NV_, E_,
      (long long)H_*NL_*NV_, (long long)NL_*NV_,
      (long long)H_*D_*NV_, (long long)D_*NV_,
      (long long)NL_*E_, D_, H_, 1.0f);

  // 9) output projections (fp32 out, +bias) straight into d_out — full tiles
  k_gemm<float, true, true><<<dim3(VD_/128, (B_*NV_)/128, 1), b256, 0, stream>>>(
      ovB, Wt_v2o, b_v2o, outV, B_*NV_, VD_, E_, E_, E_, VD_, 0,0,0,0,0,0, 1, 1.0f);
  k_gemm<float, true, true><<<dim3(LD_/128, (B_*NL_)/128, 1), b256, 0, stream>>>(
      olB, Wt_l2o, b_l2o, outL, B_*NL_, LD_, E_, E_, E_, LD_, 0,0,0,0,0,0, 1, 1.0f);
}
